// KgAdapterRGAT_71442486002205
// MI455X (gfx1250) — matrix operand order repaired and checked by reading, not compiled, not run on hardware
//
#include <hip/hip_runtime.h>

// RGAT (2-layer, R=16 relations, D=H=OUT=64) for gfx1250 / MI455X.
// v4: - fragment-packed bf16 operands -> pure b128-load + v_wmma inner loop
//     - deferred softmax normalization: one fused edge pass (exp + denom +
//       unnormalized feature scatter), divide once per node in post
//     - post fuses /denom + bias + relu + A-fragment repack (h buffer gone)
//     - bf16 [N,R,64] projection table RT-cached: 102 MB fits the 192 MB L2,
//       so ~820 MB of random edge gathers mostly hit L2, not HBM.

typedef __attribute__((ext_vector_type(16))) __bf16 v16bf;
typedef __attribute__((ext_vector_type(8)))  float  v8f;

#define DHID 64

// ---------- ordered-float <-> uint mapping for atomic segment-max ----------
__device__ __forceinline__ unsigned mapOrderedF32(float f) {
  unsigned u = __float_as_uint(f);
  return (u & 0x80000000u) ? ~u : (u | 0x80000000u);
}
__device__ __forceinline__ float unmapOrderedF32(unsigned m) {
  unsigned u = (m & 0x80000000u) ? (m ^ 0x80000000u) : ~m;
  return __uint_as_float(u);
}

// ---------- init: zero h accumulator, denom, running max ----------
__global__ void k_init(float* __restrict__ hacc, float* __restrict__ denom,
                       unsigned* __restrict__ nmax, int nNodes) {
  int i = blockIdx.x * blockDim.x + threadIdx.x;
  if (i < nNodes * DHID) hacc[i] = 0.f;
  if (i < nNodes) { denom[i] = 0.f; nmax[i] = 0u; }  // 0u == mapped "-inf" sentinel
}

// ---------- pack fp32 weights [R,64,64] -> bf16 B-fragment order ----------
// Fragment (r, kc in 2, t in 4, lane in 32) holds 16 bf16 (32B contiguous):
//   element i<8 -> W[r][kc*32 + (lane>>4)*8 + i][t*16 + (lane&15)], i>=8 -> k + 8
__global__ void k_pack_w(const float* __restrict__ W, __bf16* __restrict__ Wp, int nRel) {
  const int idx = blockIdx.x * blockDim.x + threadIdx.x;
  if (idx >= nRel * 256) return;          // 2*4*32 fragments per relation
  const int lane = idx & 31;
  const int t    = (idx >> 5) & 3;
  const int kc   = (idx >> 7) & 1;
  const int r    = idx >> 8;
  const int col  = t * 16 + (lane & 15);
  const int kb   = kc * 32 + (lane >> 4) * 8;
  const float* __restrict__ wr = W + (size_t)r * DHID * DHID;
  v16bf frag;
  #pragma unroll
  for (int i = 0; i < 16; ++i) frag[i] = (__bf16)wr[(size_t)(kb + i + (i >= 8 ? 8 : 0)) * DHID + col];
  *(v16bf*)(Wp + (size_t)idx * 16) = frag;
}

// ---------- pack fp32 activations [N,64] -> bf16 A-fragment order ----------
// Fragment (tile, kc in 2, lane in 32) holds 16 bf16 (32B contiguous):
//   row = tile*16 + (lane&15), hi = lane>>4
//   element i<8  -> X[row][kc*32 + hi*8 + i]
//   element i>=8 -> X[row][kc*32 + 16 + hi*8 + (i-8)]        (ISA 7.12.2 A layout)
__global__ void k_pack_x(const float* __restrict__ X, __bf16* __restrict__ Ap, int nNodes) {
  const int nTiles = (nNodes + 15) / 16;
  const int idx = blockIdx.x * blockDim.x + threadIdx.x;
  if (idx >= nTiles * 64) return;         // 2*32 fragments per tile
  const int lane = idx & 31;
  const int kc   = (idx >> 5) & 1;
  const int tile = idx >> 6;
  const int hi   = lane >> 4;
  int row = tile * 16 + (lane & 15);
  if (row >= nNodes) row = nNodes - 1;    // clamp (harmless duplicate)
  const float* __restrict__ xrow = X + (size_t)row * DHID + kc * 32;
  v16bf frag;
  #pragma unroll
  for (int i = 0; i < 8; ++i) frag[i]     = (__bf16)xrow[hi * 8 + i];
  #pragma unroll
  for (int i = 0; i < 8; ++i) frag[i + 8] = (__bf16)xrow[16 + hi * 8 + i];
  *(v16bf*)(Ap + (size_t)idx * 16) = frag;
}

// ---------- per-relation projection: OutAll[n,r,:] = X[n,:] @ W[r]  (bf16 out) ----------
// One wave computes a 16(M=node) x 64(H) tile for one relation r from packed frags.
// NR: compile-time relation count (0 = use runtime nRelRt).
template <int NR>
__global__ __launch_bounds__(256)
void k_proj(const __bf16* __restrict__ Ap, const __bf16* __restrict__ Wp,
            __bf16* __restrict__ OutAll, int nNodes, int nRelRt) {
  const int nr = NR ? NR : nRelRt;
  const int lane = threadIdx.x & 31;
  const int wave = threadIdx.x >> 5;
  const int r = blockIdx.y * 8 + wave;
  if (r >= nr) return;                    // wave-uniform: EXEC stays all-ones
  const int tile = blockIdx.x;
  const int node0 = tile * 16;
  const int row = lane & 15;
  const int hi  = lane >> 4;

  v8f zero;
  #pragma unroll
  for (int i = 0; i < 8; ++i) zero[i] = 0.f;
  v8f acc[4] = {zero, zero, zero, zero};

  #pragma unroll
  for (int kc = 0; kc < 2; ++kc) {
    const v16bf a = *(const v16bf*)(Ap + ((size_t)(tile * 2 + kc) * 32 + lane) * 16);
    #pragma unroll
    for (int t = 0; t < 4; ++t) {
      const v16bf b = *(const v16bf*)(Wp + ((size_t)((r * 2 + kc) * 4 + t) * 32 + lane) * 16);
      acc[t] = __builtin_amdgcn_wmma_f32_16x16x32_bf16(
          false, a, false, b, (short)0, acc[t], false, false);
    }
  }
  // C/D layout: VGPR v -> M = v + 8*hi, N = lane&15  (ISA 7.12.2)
  const size_t nodeStride = (size_t)nr * DHID;
  if (node0 + 15 < nNodes) {
    // full tile: straight-line stores, single base + immediate offsets
    __bf16* __restrict__ ob =
        OutAll + ((size_t)(node0 + hi * 8) * nr + r) * DHID + row;
    #pragma unroll
    for (int v = 0; v < 8; ++v)
      #pragma unroll
      for (int t = 0; t < 4; ++t)
        ob[(size_t)v * nodeStride + t * 16] = (__bf16)acc[t][v];
  } else {
    #pragma unroll
    for (int t = 0; t < 4; ++t) {
      const int col = t * 16 + row;
      #pragma unroll
      for (int v = 0; v < 8; ++v) {
        const int node = node0 + hi * 8 + v;
        if (node < nNodes)
          OutAll[((size_t)node * nr + r) * DHID + col] = (__bf16)acc[t][v];
      }
    }
  }
}

// ---------- attention logits: wave per edge, gather 2 rows, dual dot, leaky-relu ----------
template <int NR>
__global__ __launch_bounds__(256)
void k_logits(const __bf16* __restrict__ OutAll, const int* __restrict__ srcI,
              const int* __restrict__ dstI, const int* __restrict__ etI,
              const float* __restrict__ q, const float* __restrict__ kv,
              float* __restrict__ alphaRaw, unsigned* __restrict__ nmax,
              int nEdges, int nRelRt) {
  const int nr = NR ? NR : nRelRt;
  const int lane = threadIdx.x & 31;
  const int e = blockIdx.x * (blockDim.x >> 5) + (threadIdx.x >> 5);
  if (e >= nEdges) return;                // wave-uniform
  const int s = srcI[e], d = dstI[e], et = etI[e];
  const __bf16* oi = OutAll + ((size_t)d * nr + et) * DHID;
  const __bf16* oj = OutAll + ((size_t)s * nr + et) * DHID;
  const int c = lane * 2;
  float p = (float)oi[c] * q[c] + (float)oi[c + 1] * q[c + 1]
          + (float)oj[c] * kv[c] + (float)oj[c + 1] * kv[c + 1];
  #pragma unroll
  for (int off = 16; off > 0; off >>= 1) p += __shfl_xor(p, off, 32);
  if (lane == 0) {
    const float v = p > 0.f ? p : 0.2f * p;   // leaky_relu(., 0.2)
    alphaRaw[e] = v;
    atomicMax(nmax + d, mapOrderedF32(v));
  }
}

// ---------- fused edge pass: exp, denom, unnormalized feature scatter ----------
// segment_sum(alpha*out_j) == segment_sum(ealpha*out_j) / denom  (denom uniform
// per segment), so the division is deferred to k_post.
template <int NR>
__global__ __launch_bounds__(256)
void k_accum(const __bf16* __restrict__ OutAll, const int* __restrict__ srcI,
             const int* __restrict__ dstI, const int* __restrict__ etI,
             const float* __restrict__ alphaRaw, const unsigned* __restrict__ nmax,
             float* __restrict__ ealpha, float* __restrict__ denom,
             float* __restrict__ hacc, int nEdges, int nRelRt) {
  const int nr = NR ? NR : nRelRt;
  const int lane = threadIdx.x & 31;
  const int e = blockIdx.x * (blockDim.x >> 5) + (threadIdx.x >> 5);
  if (e >= nEdges) return;                // wave-uniform
  const int s = srcI[e], d = dstI[e], et = etI[e];
  const float ea = __expf(alphaRaw[e] - unmapOrderedF32(nmax[d]));
  if (lane == 0) {
    ealpha[e] = ea;
    atomicAdd(denom + d, ea);
  }
  const __bf16* oj = OutAll + ((size_t)s * nr + et) * DHID;
  float* hd = hacc + (size_t)d * DHID;
  const int c = lane * 2;
  atomicAdd(hd + c,     ea * (float)oj[c]);
  atomicAdd(hd + c + 1, ea * (float)oj[c + 1]);
}

// ---------- normalized attention weights output (layer 2 tuple slot) ----------
__global__ void k_alpha_out(const float* __restrict__ ealpha, const int* __restrict__ dstI,
                            const float* __restrict__ denom, float* __restrict__ alphaOut,
                            int nEdges) {
  const int e = blockIdx.x * blockDim.x + threadIdx.x;
  if (e >= nEdges) return;
  alphaOut[e] = ealpha[e] / (denom[dstI[e]] + 1e-16f);
}

// ---------- post: h = relu(hacc/denom + bias), emitted directly as packed A-frags ----
__global__ void k_post(const float* __restrict__ hacc, const float* __restrict__ denom,
                       const float* __restrict__ bias, __bf16* __restrict__ Ap,
                       int nNodes) {
  const int nTiles = (nNodes + 15) / 16;
  const int idx = blockIdx.x * blockDim.x + threadIdx.x;
  if (idx >= nTiles * 64) return;
  const int lane = idx & 31;
  const int kc   = (idx >> 5) & 1;
  const int tile = idx >> 6;
  const int hi   = lane >> 4;
  int row = tile * 16 + (lane & 15);
  if (row >= nNodes) row = nNodes - 1;
  const float inv = 1.f / (denom[row] + 1e-16f);
  const float* __restrict__ hrow = hacc + (size_t)row * DHID + kc * 32;
  const float* __restrict__ brow = bias + kc * 32;
  v16bf frag;
  #pragma unroll
  for (int i = 0; i < 8; ++i) {
    const float v = hrow[hi * 8 + i] * inv + brow[hi * 8 + i];
    frag[i] = (__bf16)(v > 0.f ? v : 0.f);
  }
  #pragma unroll
  for (int i = 0; i < 8; ++i) {
    const float v = hrow[16 + hi * 8 + i] * inv + brow[16 + hi * 8 + i];
    frag[i + 8] = (__bf16)(v > 0.f ? v : 0.f);
  }
  *(v16bf*)(Ap + (size_t)idx * 16) = frag;
}

// ---------- final linear: Out = H @ Wl + bl  (packed frags, one wave / 16-node tile) ----------
__global__ __launch_bounds__(32)
void k_linear(const __bf16* __restrict__ Ap, const __bf16* __restrict__ Wp,
              const float* __restrict__ bl, float* __restrict__ Out, int nNodes) {
  const int lane = threadIdx.x & 31;
  const int tile = blockIdx.x;
  const int node0 = tile * 16;
  const int row = lane & 15;
  const int hi  = lane >> 4;

  v8f zero;
  #pragma unroll
  for (int i = 0; i < 8; ++i) zero[i] = 0.f;
  v8f acc[4] = {zero, zero, zero, zero};

  #pragma unroll
  for (int kc = 0; kc < 2; ++kc) {
    const v16bf a = *(const v16bf*)(Ap + ((size_t)(tile * 2 + kc) * 32 + lane) * 16);
    #pragma unroll
    for (int t = 0; t < 4; ++t) {
      const v16bf b = *(const v16bf*)(Wp + ((size_t)(kc * 4 + t) * 32 + lane) * 16);
      acc[t] = __builtin_amdgcn_wmma_f32_16x16x32_bf16(
          false, a, false, b, (short)0, acc[t], false, false);
    }
  }
  if (node0 + 15 < nNodes) {
    float* __restrict__ ob = Out + (size_t)(node0 + hi * 8) * DHID + row;
    #pragma unroll
    for (int v = 0; v < 8; ++v)
      #pragma unroll
      for (int t = 0; t < 4; ++t)
        ob[v * DHID + t * 16] = acc[t][v] + bl[t * 16 + row];
  } else {
    #pragma unroll
    for (int t = 0; t < 4; ++t) {
      const int col = t * 16 + row;
      #pragma unroll
      for (int v = 0; v < 8; ++v) {
        const int node = node0 + hi * 8 + v;
        if (node < nNodes)
          Out[(size_t)node * DHID + col] = acc[t][v] + bl[col];
      }
    }
  }
}

extern "C" void kernel_launch(void* const* d_in, const int* in_sizes, int n_in,
                              void* d_out, int out_size, void* d_ws, size_t ws_size,
                              hipStream_t stream) {
  const float* x        = (const float*)d_in[0];
  const int*   edge_idx = (const int*)d_in[1];   // [2,E] row-major
  const int*   edge_ty  = (const int*)d_in[2];
  const float* w1       = (const float*)d_in[3];
  const float* q1       = (const float*)d_in[4];
  const float* k1       = (const float*)d_in[5];
  const float* b1       = (const float*)d_in[6];
  const float* w2       = (const float*)d_in[7];
  const float* q2       = (const float*)d_in[8];
  const float* k2       = (const float*)d_in[9];
  const float* b2       = (const float*)d_in[10];
  const float* lin_w    = (const float*)d_in[11];
  const float* lin_b    = (const float*)d_in[12];

  const int N = in_sizes[0] / DHID;
  const int E = in_sizes[1] / 2;
  const int R = in_sizes[3] / (DHID * DHID);

  const int* srcI = edge_idx;        // edge_index[0]
  const int* dstI = edge_idx + E;    // edge_index[1]

  // workspace carve-up (~130 MB)
  char* ws = (char*)d_ws;
  auto al = [](size_t v) { return (v + 255) & ~(size_t)255; };
  size_t off = 0;
  __bf16*   outAll   = (__bf16*)(ws + off);   off = al(off + (size_t)N * R * DHID * sizeof(__bf16));
  float*    hacc     = (float*)(ws + off);    off = al(off + (size_t)N * DHID * sizeof(float));
  float*    alphaRaw = (float*)(ws + off);    off = al(off + (size_t)E * sizeof(float));
  float*    ealpha   = (float*)(ws + off);    off = al(off + (size_t)E * sizeof(float));
  unsigned* nmax     = (unsigned*)(ws + off); off = al(off + (size_t)N * sizeof(unsigned));
  float*    denom    = (float*)(ws + off);    off = al(off + (size_t)N * sizeof(float));
  const int nTiles   = (N + 15) / 16;
  __bf16*   apack    = (__bf16*)(ws + off);   off = al(off + (size_t)nTiles * 64 * 16 * sizeof(__bf16));
  __bf16*   wpack1   = (__bf16*)(ws + off);   off = al(off + (size_t)R * 256 * 16 * sizeof(__bf16));
  __bf16*   wpack2   = (__bf16*)(ws + off);   off = al(off + (size_t)R * 256 * 16 * sizeof(__bf16));
  __bf16*   linpack  = (__bf16*)(ws + off);   off = al(off + (size_t)256 * 16 * sizeof(__bf16));

  float* outPtr   = (float*)d_out;                              // tuple slot 0: [N,64]
  float* alphaOut = outPtr + (size_t)N * DHID + (size_t)2 * E;  // tuple slot 2: [E,1]

  dim3 projGrid(nTiles, (R + 7) / 8);
  const int edgeWaveBlocks = (E + 7) / 8;       // 8 waves/block, wave per edge
  const int initBlocks = (N * DHID + 255) / 256;
  const int fragBlocks = (nTiles * 64 + 255) / 256;
  const bool r16 = (R == 16);

  auto proj = [&](const __bf16* wp) {
    if (r16) k_proj<16><<<projGrid, 256, 0, stream>>>(apack, wp, outAll, N, R);
    else     k_proj<0 ><<<projGrid, 256, 0, stream>>>(apack, wp, outAll, N, R);
  };
  auto logits = [&](const float* q, const float* kv) {
    if (r16) k_logits<16><<<edgeWaveBlocks, 256, 0, stream>>>(outAll, srcI, dstI, edge_ty,
                                                              q, kv, alphaRaw, nmax, E, R);
    else     k_logits<0 ><<<edgeWaveBlocks, 256, 0, stream>>>(outAll, srcI, dstI, edge_ty,
                                                              q, kv, alphaRaw, nmax, E, R);
  };
  auto accum = [&]() {
    if (r16) k_accum<16><<<edgeWaveBlocks, 256, 0, stream>>>(outAll, srcI, dstI, edge_ty,
                                                             alphaRaw, nmax, ealpha, denom,
                                                             hacc, E, R);
    else     k_accum<0 ><<<edgeWaveBlocks, 256, 0, stream>>>(outAll, srcI, dstI, edge_ty,
                                                             alphaRaw, nmax, ealpha, denom,
                                                             hacc, E, R);
  };

  // ---- pack all weights once ----
  k_pack_w<<<(R * 256 + 255) / 256, 256, 0, stream>>>(w1, wpack1, R);
  k_pack_w<<<(R * 256 + 255) / 256, 256, 0, stream>>>(w2, wpack2, R);
  k_pack_w<<<1, 256, 0, stream>>>(lin_w, linpack, 1);

  // ---- Layer 1 ----
  k_init<<<initBlocks, 256, 0, stream>>>(hacc, denom, nmax, N);
  k_pack_x<<<fragBlocks, 256, 0, stream>>>(x, apack, N);
  proj(wpack1);
  logits(q1, k1);
  accum();
  k_post<<<fragBlocks, 256, 0, stream>>>(hacc, denom, b1, apack, N);  // -> layer-2 input

  // ---- Layer 2 ----
  k_init<<<initBlocks, 256, 0, stream>>>(hacc, denom, nmax, N);
  proj(wpack2);
  logits(q2, k2);
  accum();
  k_alpha_out<<<(E + 255) / 256, 256, 0, stream>>>(ealpha, dstI, denom, alphaOut, E);
  k_post<<<fragBlocks, 256, 0, stream>>>(hacc, denom, b2, apack, N);  // -> linear input

  // ---- Final linear ----
  k_linear<<<nTiles, 32, 0, stream>>>(apack, linpack, lin_b, outPtr, N);

  // tuple slot 1: edge_index passthrough (bit copy, capture-legal d2d async)
  hipMemcpyAsync(outPtr + (size_t)N * DHID, edge_idx, (size_t)2 * E * sizeof(int),
                 hipMemcpyDeviceToDevice, stream);
}
